// WeightedCCEFocalTverskyLossWithSoftmax_64836826300731
// MI455X (gfx1250) — matrix operand order repaired and checked by reading, not compiled, not run on hardware
//
#include <hip/hip_runtime.h>
#include <math.h>

typedef __attribute__((ext_vector_type(16))) _Float16 v16h;
typedef __attribute__((ext_vector_type(8)))  float    v8f;

#define WAVES_PER_BLOCK 8
#define BLOCK 256
#define GRID 512

static constexpr float TV_ALPHA    = 0.7f;
static constexpr float TV_BETA     = 0.3f;   // 1 - alpha
static constexpr float FOCAL_GAMMA = 0.75f;
static constexpr float EPS_        = 1e-8f;
static constexpr float CCE_W       = 0.1f;
static constexpr float DICE_W      = 1.0f;

// ---------------------------------------------------------------------------
// Kernel 0: zero the 256 confusion-matrix bins in workspace (ws is poisoned).
// ---------------------------------------------------------------------------
__global__ void cm_init_kernel(float* __restrict__ cm) {
  cm[threadIdx.x] = 0.0f;   // launched with 256 threads
}

// ---------------------------------------------------------------------------
// Kernel 1: streaming pass. Each lane owns one sample per chunk of 32.
//   - softmax stats, argmax(pred), argmax(gt one-hot), CE term
//   - confusion matrix as sum of rank-1 one-hot outer products, computed as
//     a 16x16x32 f16 WMMA per wave-chunk: CM += A(16x32,gt-onehot^T) x
//     B(32x16,pred-onehot). f32 accumulator holds exact integer counts.
// ---------------------------------------------------------------------------
__global__ __launch_bounds__(BLOCK)
void wcce_ftl_main_kernel(const float* __restrict__ pred,
                          const float* __restrict__ gt,
                          const float* __restrict__ wts,
                          float* __restrict__ cm_out,      // [256] global, atomic f32 (exact ints)
                          float* __restrict__ ce_partial,  // [gridDim.x]
                          int n) {
  __shared__ float    sw[16];
  __shared__ float    lcm[256];
  __shared__ unsigned stage[WAVES_PER_BLOCK][32];  // per-wave (gt | pred<<8)
  __shared__ float    ce_wave[WAVES_PER_BLOCK];

  const int tid = threadIdx.x;
  if (tid < 16) sw[tid] = wts[tid];
  lcm[tid] = 0.0f;                               // BLOCK == 256
  __syncthreads();

  const int  lane = tid & 31;
  const int  wave = tid >> 5;
  const int  half = lane >> 4;                   // 0: lanes 0-15, 1: lanes 16-31
  const int  Lm   = lane & 15;
  const long totalWaves = (long)gridDim.x * WAVES_PER_BLOCK;
  const long wgid       = (long)blockIdx.x * WAVES_PER_BLOCK + wave;

  v8f   cmacc  = {};                             // 16x16 CM partial (WMMA C/D layout)
  float ce_acc = 0.0f;

  // full 32-sample chunks only (N = 4194304 is a multiple of 32)
  for (long s0 = wgid * 32; s0 + 31 < (long)n; s0 += totalWaves * 32) {
    const long s = s0 + lane;

    // ---- load 16 logits (4x global_load_b128) ----
    const float4* xr = (const float4*)(pred + s * 16);
    float4 q0 = xr[0], q1 = xr[1], q2 = xr[2], q3 = xr[3];
    float x[16] = {q0.x,q0.y,q0.z,q0.w, q1.x,q1.y,q1.z,q1.w,
                   q2.x,q2.y,q2.z,q2.w, q3.x,q3.y,q3.z,q3.w};

    // ---- max + first-occurrence argmax (matches jnp.argmax) ----
    float m = x[0]; int pcat = 0;
#pragma unroll
    for (int j = 1; j < 16; ++j) { if (x[j] > m) { m = x[j]; pcat = j; } }

    // ---- softmax: e_j = exp(x_j - m), s = sum e ----
    float ssum = 0.0f;
#pragma unroll
    for (int j = 0; j < 16; ++j) { x[j] = __expf(x[j] - m); ssum += x[j]; }
    const float inv_s = 1.0f / ssum;
    const float m2    = inv_s;                   // max_j p_j == exp(0)/ssum exactly

    // ---- ground-truth row: one-hot index + selected prob ----
    const float4* gr = (const float4*)(gt + s * 16);
    float4 h0 = gr[0], h1 = gr[1], h2 = gr[2], h3 = gr[3];
    float g[16] = {h0.x,h0.y,h0.z,h0.w, h1.x,h1.y,h1.z,h1.w,
                   h2.x,h2.y,h2.z,h2.w, h3.x,h3.y,h3.z,h3.w};
    float eg = x[0]; int gcat = 0;
#pragma unroll
    for (int j = 1; j < 16; ++j) {
      bool c = g[j] > 0.5f;                      // one-hot: exactly one element == 1.0
      eg   = c ? x[j] : eg;
      gcat = c ? j    : gcat;
    }

    // ---- second log-softmax (reference applies log_softmax to softmax out) ----
    float s2 = 0.0f;
#pragma unroll
    for (int j = 0; j < 16; ++j) s2 += __expf(x[j] * inv_s - m2);
    const float pg = eg * inv_s;
    // ce = -w[g] * (p_g - m2 - log(s2))
    ce_acc += sw[gcat] * (m2 + __logf(s2) - pg);

    // ---- stage indices for this wave's 32 samples, build WMMA tiles ----
    stage[wave][lane] = (unsigned)gcat | ((unsigned)pcat << 8);
    const unsigned* st = stage[wave];            // same-wave RAW: DScnt-ordered

    v16h A, B;
    // A (16x32 f16, gt one-hot transposed): row M = Lm
    //   lanes 0-15 : V0..V3 -> K=0..7,   V4..V7 -> K=8..15
    //   lanes 16-31: V0..V3 -> K=16..23, V4..V7 -> K=24..31
    const int base0 = half ? 16 : 0;
    const int base1 = half ? 24 : 8;
#pragma unroll
    for (int v = 0; v < 4; ++v) {
      unsigned w0 = st[base0 + 2*v], w1 = st[base0 + 2*v + 1];
      A[2*v]     = ((w0 & 0xFFu) == (unsigned)Lm) ? (_Float16)1.0f : (_Float16)0.0f;
      A[2*v + 1] = ((w1 & 0xFFu) == (unsigned)Lm) ? (_Float16)1.0f : (_Float16)0.0f;
    }
#pragma unroll
    for (int v = 0; v < 4; ++v) {
      unsigned w0 = st[base1 + 2*v], w1 = st[base1 + 2*v + 1];
      A[8 + 2*v]     = ((w0 & 0xFFu) == (unsigned)Lm) ? (_Float16)1.0f : (_Float16)0.0f;
      A[8 + 2*v + 1] = ((w1 & 0xFFu) == (unsigned)Lm) ? (_Float16)1.0f : (_Float16)0.0f;
    }
    // B (32x16 f16, pred one-hot): column N = Lm
    //   lanes 0-15 hold K=0..15, lanes 16-31 hold K=16..31, 2 K per VGPR
    const int kb = half ? 16 : 0;
#pragma unroll
    for (int v = 0; v < 8; ++v) {
      unsigned w0 = st[kb + 2*v], w1 = st[kb + 2*v + 1];
      B[2*v]     = (((w0 >> 8) & 0xFFu) == (unsigned)Lm) ? (_Float16)1.0f : (_Float16)0.0f;
      B[2*v + 1] = (((w1 >> 8) & 0xFFu) == (unsigned)Lm) ? (_Float16)1.0f : (_Float16)0.0f;
    }
    // CM += A x B  (counts are integers < 2^24 -> exact in f32)
    cmacc = __builtin_amdgcn_wmma_f32_16x16x32_f16(
        false, A, false, B, (short)0, cmacc, false, false);
  }

  // ---- wave-reduce CE, then per-block fixed-order sum -> workspace ----
#pragma unroll
  for (int off = 16; off > 0; off >>= 1)
    ce_acc += __shfl_xor(ce_acc, off, 32);
  if (lane == 0) ce_wave[wave] = ce_acc;

  // ---- merge per-wave CM tiles into LDS (integer-valued f32 -> exact) ----
#pragma unroll
  for (int v = 0; v < 8; ++v) {
    int M = half ? (v + 8) : v;                  // C/D layout: lanes 16-31 hold M=8..15
    atomicAdd(&lcm[M * 16 + Lm], cmacc[v]);
  }
  __syncthreads();

  if (tid == 0) {
    float cs = 0.0f;
    for (int i = 0; i < WAVES_PER_BLOCK; ++i) cs += ce_wave[i];
    ce_partial[blockIdx.x] = cs;
  }
  atomicAdd(&cm_out[tid], lcm[tid]);             // 256 exact-integer f32 atomics per block
}

// ---------------------------------------------------------------------------
// Kernel 2: single block. Fixed-order CE reduction + Tversky/focal + combine.
// ---------------------------------------------------------------------------
__global__ __launch_bounds__(256)
void wcce_ftl_finalize_kernel(const float* __restrict__ cm,
                              const float* __restrict__ ce_partial, int nblocks,
                              const float* __restrict__ wts,
                              float* __restrict__ out, float n_samples) {
  __shared__ float red[256];
  __shared__ float fsum[16], wden[16];
  const int t = threadIdx.x;

  float s = 0.0f;
  for (int i = t; i < nblocks; i += 256) s += ce_partial[i];  // fixed order
  red[t] = s;
  __syncthreads();
#pragma unroll
  for (int off = 128; off > 0; off >>= 1) {
    if (t < off) red[t] += red[t + off];
    __syncthreads();
  }

  if (t < 16) {
    const int c = t;
    float tp = cm[c * 16 + c];
    float col = 0.0f, row = 0.0f;
    for (int r = 0; r < 16; ++r) { col += cm[r * 16 + c]; row += cm[c * 16 + r]; }
    float fp = col - tp, fn = row - tp;
    float tv = (tp + EPS_) / (tp + fp * TV_BETA + fn * TV_ALPHA + EPS_);
    fsum[c] = powf(1.0f - tv, FOCAL_GAMMA) * wts[c];
    wden[c] = wts[c] * row;                      // sum w[gt_n] == sum_c w_c * rowsum_c
  }
  __syncthreads();

  if (t == 0) {
    float fs = 0.0f, wd = 0.0f;
    for (int c = 0; c < 16; ++c) { fs += fsum[c]; wd += wden[c]; }
    float ftl = fs / wd;
    float cce = red[0] / n_samples;
    out[0] = cce * CCE_W + ftl * DICE_W;
  }
}

// ---------------------------------------------------------------------------
extern "C" void kernel_launch(void* const* d_in, const int* in_sizes, int n_in,
                              void* d_out, int out_size, void* d_ws, size_t ws_size,
                              hipStream_t stream) {
  const float* pred = (const float*)d_in[0];   // [N,16] f32
  const float* gt   = (const float*)d_in[1];   // [N,16] f32 one-hot
  const float* wts  = (const float*)d_in[2];   // [16]   f32
  const int n = in_sizes[0] / 16;              // N rows

  float* cm         = (float*)d_ws;            // 256 f32
  float* ce_partial = cm + 256;                // GRID f32

  cm_init_kernel<<<1, 256, 0, stream>>>(cm);
  wcce_ftl_main_kernel<<<GRID, BLOCK, 0, stream>>>(pred, gt, wts, cm, ce_partial, n);
  wcce_ftl_finalize_kernel<<<1, 256, 0, stream>>>(cm, ce_partial, GRID, wts,
                                                  (float*)d_out, (float)n);
}
